// VectorQuantizer_13572096655575
// MI455X (gfx1250) — matrix operand order repaired and checked by reading, not compiled, not run on hardware
//
#include <hip/hip_runtime.h>

#define N_VEC   32768      // 32*32*32 vectors
#define DIM     256
#define K_CB    1024
#define COMMIT  0.25f

// LDS staging: chunks of 4 column tiles (64 codebook entries)
#define CHUNK_ENTRIES 64
#define NCHUNK        (K_CB / CHUNK_ENTRIES)      // 16
#define ROW_ELEMS     264                          // 256 + 8 f16 pad (528B row)
#define CHUNK_ELEMS   (CHUNK_ENTRIES * ROW_ELEMS)  // 16896 f16 = 33792 B
#define SEGS_PER_THR  8                            // 2048 16B-segments / 256 thr

typedef __attribute__((ext_vector_type(16))) _Float16 v16h;
typedef __attribute__((ext_vector_type(8)))  _Float16 v8h;
typedef __attribute__((ext_vector_type(4)))  _Float16 v4h;
typedef __attribute__((ext_vector_type(8)))  float    v8f;
typedef __attribute__((ext_vector_type(4)))  int      v4i;

typedef __attribute__((address_space(1))) v4i* gptr_v4i;   // global int4*
typedef __attribute__((address_space(3))) v4i* lptr_v4i;   // LDS int4*

union V16H { v16h v; v8h h[2]; };

#if __has_builtin(__builtin_amdgcn_global_load_async_to_lds_b128)
#define HAS_ASYNC_LDS 1
#else
#define HAS_ASYNC_LDS 0
#endif
#if __has_builtin(__builtin_amdgcn_s_wait_asynccnt)
#define WAIT_ASYNC(n) __builtin_amdgcn_s_wait_asynccnt(n)
#elif HAS_ASYNC_LDS
#define WAIT_ASYNC(n) asm volatile("s_wait_asynccnt %0" ::"i"(n) : "memory")
#else
#define WAIT_ASYNC(n)
#endif

// ---------------------------------------------------------------------------
// Pass 1: f32 -> f16 conversion of z and codebook into workspace (vectorized)
// ---------------------------------------------------------------------------
__global__ void vq_prep(const float* __restrict__ z, const float* __restrict__ cb,
                        _Float16* __restrict__ zh, _Float16* __restrict__ cbh) {
    const int tid = blockIdx.x * blockDim.x + threadIdx.x;
    const int stride = gridDim.x * blockDim.x;
    const float4* z4  = (const float4*)z;
    const float4* cb4 = (const float4*)cb;
    v4h* zh4  = (v4h*)zh;
    v4h* cbh4 = (v4h*)cbh;
    for (int j = tid; j < (N_VEC * DIM) / 4; j += stride) {
        float4 f = z4[j];
        v4h h; h.x = (_Float16)f.x; h.y = (_Float16)f.y;
               h.z = (_Float16)f.z; h.w = (_Float16)f.w;
        zh4[j] = h;
    }
    for (int j = tid; j < (K_CB * DIM) / 4; j += stride) {
        float4 f = cb4[j];
        v4h h; h.x = (_Float16)f.x; h.y = (_Float16)f.y;
               h.z = (_Float16)f.z; h.w = (_Float16)f.w;
        cbh4[j] = h;
    }
}

// ---------------------------------------------------------------------------
// Pass 2: 0.5 * ||e_k||^2 per codebook entry (one wave per entry)
// ---------------------------------------------------------------------------
__global__ void vq_norms(const float* __restrict__ cb, float* __restrict__ nh) {
    const int wave = (blockIdx.x * blockDim.x + threadIdx.x) >> 5;
    const int lane = threadIdx.x & 31;
    if (wave >= K_CB) return;
    const float* r = cb + (size_t)wave * DIM;
    float s = 0.f;
    #pragma unroll
    for (int j = 0; j < DIM / 32; ++j) {
        float v = r[lane + j * 32];
        s += v * v;
    }
    #pragma unroll
    for (int m = 16; m; m >>= 1) s += __shfl_xor(s, m, 32);
    if (lane == 0) nh[wave] = 0.5f * s;
}

// ---------------------------------------------------------------------------
// Pass 3: main VQ kernel. 8 waves/block, one 16-row tile per wave.
// Codebook staged through LDS with double-buffered async global->LDS copies.
// ---------------------------------------------------------------------------
__global__ void __launch_bounds__(256)
vq_main(const float* __restrict__ z, const float* __restrict__ cb,
        const _Float16* __restrict__ zh, const _Float16* __restrict__ cbh,
        const float* __restrict__ nh,
        float* __restrict__ outq, float* __restrict__ enc,
        float* __restrict__ partial) {
    __shared__ __align__(16) _Float16 sbuf[2 * CHUNK_ELEMS];   // 66 KB

    const int tid  = threadIdx.x;
    const int lane = tid & 31;
    const int wave = tid >> 5;
    const int l16  = lane & 15;
    const int hi   = lane >> 4;              // 0: lanes 0-15, 1: lanes 16-31
    const int rt   = blockIdx.x * 8 + wave;  // row tile id [0, 2048)
    const int n0   = rt * 16;

    // cooperative async stage of one 64-entry codebook chunk into LDS buffer
    auto issue_chunk = [&](int chunk, int buf) {
        #pragma unroll
        for (int i = 0; i < SEGS_PER_THR; ++i) {
            const int seg   = tid + i * 256;     // 16B segment id [0, 2048)
            const int entry = seg >> 5;          // [0, 64)
            const int s     = seg & 31;          // 16B segment within 512B row
            const _Float16* g = cbh + ((size_t)(chunk * CHUNK_ENTRIES + entry)) * DIM + s * 8;
            _Float16* l = sbuf + (size_t)buf * CHUNK_ELEMS + entry * ROW_ELEMS + s * 8;
#if HAS_ASYNC_LDS
            __builtin_amdgcn_global_load_async_to_lds_b128(
                (gptr_v4i)g, (lptr_v4i)l, 0, 0);
#else
            *(float4*)l = *(const float4*)g;     // sync fallback via VGPRs
#endif
        }
    };

    // ---- A tiles: 16 rows x 256 K of f16, contiguous K per lane half (k = 16h + s) ----
    V16H A[8];
    {
        const _Float16* zr = zh + (size_t)(n0 + l16) * DIM + hi * 16;
        #pragma unroll
        for (int ks = 0; ks < 8; ++ks) {
            A[ks].h[0] = *(const v8h*)(zr + ks * 32);
            A[ks].h[1] = *(const v8h*)(zr + ks * 32 + 8);
        }
    }

    float best[8];
    int   bcol[8];
    #pragma unroll
    for (int r = 0; r < 8; ++r) { best[r] = -3.4e38f; bcol[r] = 0; }

    issue_chunk(0, 0);

    #pragma unroll 1
    for (int c = 0; c < NCHUNK; ++c) {
        const int buf = c & 1;
        if (c + 1 < NCHUNK) {
            issue_chunk(c + 1, buf ^ 1);   // prefetch next chunk
            WAIT_ASYNC(SEGS_PER_THR);      // chunk c done; c+1 still in flight
        } else {
            WAIT_ASYNC(0);
        }
        __syncthreads();                   // chunk c visible to all waves

        #pragma unroll
        for (int t = 0; t < CHUNK_ENTRIES / 16; ++t) {       // 4 column tiles
            const int ct  = c * (CHUNK_ENTRIES / 16) + t;
            const int col = ct * 16 + l16;
            const float cinit = -nh[col];                    // -0.5*||e||^2
            v8f cc = { cinit, cinit, cinit, cinit, cinit, cinit, cinit, cinit };
            // B (32x16 f16): lane holds 16 contiguous K of its column, padded
            // row stride 528B => half-group lanes start at banks 0,4,...,60
            const _Float16* br = sbuf + (size_t)buf * CHUNK_ELEMS
                                 + (t * 16 + l16) * ROW_ELEMS + hi * 16;
            #pragma unroll
            for (int ks = 0; ks < 8; ++ks) {
                v16h B = *(const v16h*)(br + ks * 32);
                cc = __builtin_amdgcn_wmma_f32_16x16x32_f16(
                        false, A[ks].v, false, B, (short)0, cc, false, false);
            }
            #pragma unroll
            for (int r = 0; r < 8; ++r) {
                if (cc[r] > best[r]) { best[r] = cc[r]; bcol[r] = col; }
            }
        }
        __syncthreads();                   // all waves done reading buffer c&1
    }

    // ---- per-row argmax across the 16 lanes of each half-group ----
    #pragma unroll
    for (int m = 1; m < 16; m <<= 1) {
        #pragma unroll
        for (int r = 0; r < 8; ++r) {
            float ov = __shfl_xor(best[r], m, 32);
            int   oc = __shfl_xor(bcol[r], m, 32);
            if (ov > best[r] || (ov == best[r] && oc < bcol[r])) {
                best[r] = ov; bcol[r] = oc;
            }
        }
    }
    int idxRow[16];
    #pragma unroll
    for (int r = 0; r < 8; ++r) {
        idxRow[r]     = __shfl(bcol[r], 0, 32);
        idxRow[r + 8] = __shfl(bcol[r], 16, 32);
    }

    // ---- outputs: quantized rows, one-hot encodings, loss partial ----
    float acc = 0.f;
    #pragma unroll 1
    for (int r = 0; r < 16; ++r) {
        const int n = n0 + r;
        const int e = idxRow[r];
        const float4* crow = (const float4*)(cb   + (size_t)e * DIM);
        const float4* zrow = (const float4*)(z    + (size_t)n * DIM);
        float4*       qrow = (float4*)      (outq + (size_t)n * DIM);
        #pragma unroll
        for (int j = 0; j < 2; ++j) {
            const int t = lane + j * 32;     // 64 float4 per row
            float4 q  = crow[t];
            float4 zz = zrow[t];
            qrow[t] = q;
            float dx = q.x - zz.x, dy = q.y - zz.y;
            float dz = q.z - zz.z, dw = q.w - zz.w;
            acc += dx * dx + dy * dy + dz * dz + dw * dw;
        }
        // encodings row (1024 floats). Tuple offset makes base 4B-misaligned
        // for float4, so use coalesced b32 stores.
        float* er = enc + (size_t)n * K_CB;
        #pragma unroll
        for (int j = 0; j < K_CB / 32; ++j) {
            const int col = j * 32 + lane;
            er[col] = (col == e) ? 1.f : 0.f;
        }
    }

    // deterministic loss partial: wave shuffle-sum, then fixed-order LDS sum
    #pragma unroll
    for (int m = 16; m; m >>= 1) acc += __shfl_xor(acc, m, 32);
    __shared__ float sacc[8];
    if (lane == 0) sacc[wave] = acc;
    __syncthreads();
    if (threadIdx.x == 0) {
        float s = 0.f;
        #pragma unroll
        for (int wv = 0; wv < 8; ++wv) s += sacc[wv];
        partial[blockIdx.x] = s;
    }
}

// ---------------------------------------------------------------------------
// Pass 4: final loss = (1 + COMMIT) * mean((q - z)^2)
// ---------------------------------------------------------------------------
__global__ void vq_loss(const float* __restrict__ partial, float* __restrict__ loss_out) {
    __shared__ float s[256];
    s[threadIdx.x] = partial[threadIdx.x];
    __syncthreads();
    #pragma unroll
    for (int m = 128; m; m >>= 1) {
        if (threadIdx.x < m) s[threadIdx.x] += s[threadIdx.x + m];
        __syncthreads();
    }
    if (threadIdx.x == 0)
        *loss_out = s[0] * (1.f + COMMIT) / (float)(N_VEC * DIM);
}

// ---------------------------------------------------------------------------
extern "C" void kernel_launch(void* const* d_in, const int* in_sizes, int n_in,
                              void* d_out, int out_size, void* d_ws, size_t ws_size,
                              hipStream_t stream) {
    (void)in_sizes; (void)n_in; (void)out_size; (void)ws_size;
    const float* z  = (const float*)d_in[0];   // [32,32,32,256] f32
    const float* cb = (const float*)d_in[1];   // [1024,256] f32

    float* out  = (float*)d_out;
    float* outq = out;                                   // 8388608 floats
    float* loss = out + (size_t)N_VEC * DIM;             // 1 float
    float* enc  = loss + 1;                              // 33554432 floats

    char* ws = (char*)d_ws;
    _Float16* zh  = (_Float16*)ws;                                   // 16.78 MB
    _Float16* cbh = (_Float16*)(ws + (size_t)N_VEC * DIM * 2);       // 512 KB
    float* nh      = (float*)(ws + (size_t)N_VEC * DIM * 2
                                 + (size_t)K_CB * DIM * 2);          // 4 KB
    float* partial = nh + K_CB;                                      // 1 KB

    vq_prep <<<2048, 256, 0, stream>>>(z, cb, zh, cbh);
    vq_norms<<<K_CB / 8, 256, 0, stream>>>(cb, nh);
    vq_main <<<N_VEC / (16 * 8), 256, 0, stream>>>(z, cb, zh, cbh, nh,
                                                   outq, enc, partial);
    vq_loss <<<1, 256, 0, stream>>>(partial, loss);
}
